// _SimClrLossNaive_44985487459095
// MI455X (gfx1250) — matrix operand order repaired and checked by reading, not compiled, not run on hardware
//
#include <hip/hip_runtime.h>
#include <hip/hip_bf16.h>

typedef _Float16      v16h __attribute__((ext_vector_type(16)));
typedef _Float16      v4h  __attribute__((ext_vector_type(4)));
typedef float         v8f  __attribute__((ext_vector_type(8)));
typedef unsigned int  v4u  __attribute__((ext_vector_type(4)));

#define TWO_N   8192
#define NHALF   4096
#define DIM     256
#define TILES_M 4                         // 16-row strips per wave
#define ROWS_PER_WAVE (TILES_M * 16)      // 64
#define ROWS_PER_BLK  (8 * ROWS_PER_WAVE) // 512
#define SEGS    16
#define COL_TILES_PER_SEG ((TWO_N / 16) / SEGS)   // 32

// Scale folded into the GEMM operands: (s*a)·(s*b) = s^2 * sim = sim * (1/T) * log2(e),
// so the WMMA result feeds v_exp_f32 (base-2) directly with zero extra VALU.
#define SCALE 1.6986436798f               // sqrt(2 * log2(e))
#define LN2   0.6931471806f               // converts captured pos value back: d_pos*ln2 = sim/T

union HFrag { v16h h; v4u u[2]; };

// ---- CDNA5 async memory->LDS copy (ASYNCcnt path, bypasses VGPRs) ----------
__device__ __forceinline__ unsigned lds_low32(const void* p) {
  return (unsigned)(unsigned long long)p;   // low 32 bits of generic = LDS byte address
}
__device__ __forceinline__ void async_ld_b128(unsigned ldsaddr, const void* gaddr) {
  asm volatile("global_load_async_to_lds_b128 %0, %1, off"
               :: "v"(ldsaddr), "v"(gaddr) : "memory");
}
__device__ __forceinline__ void wait_async0() {
#if __has_builtin(__builtin_amdgcn_s_wait_asynccnt)
  __builtin_amdgcn_s_wait_asynccnt(0);
#else
  asm volatile("s_wait_asynccnt 0x0" ::: "memory");
#endif
}
__device__ __forceinline__ float fast_exp2(float x) {
#if __has_builtin(__builtin_amdgcn_exp2f)
  return __builtin_amdgcn_exp2f(x);
#else
  return exp2f(x);
#endif
}

// ---------------- Kernel 1: row-normalize fp32 -> fp16*SCALE (wave per row) -
__global__ __launch_bounds__(256) void _simclr_norm(const float* __restrict__ x,
                                                    _Float16* __restrict__ xn) {
  const int row  = (blockIdx.x * 256 + threadIdx.x) >> 5;
  const int lane = threadIdx.x & 31;
  const float4* xr = (const float4*)(x + (size_t)row * DIM);
  float4 a = xr[lane];
  float4 b = xr[lane + 32];
  float ss = a.x*a.x + a.y*a.y + a.z*a.z + a.w*a.w
           + b.x*b.x + b.y*b.y + b.z*b.z + b.w*b.w;
  #pragma unroll
  for (int off = 16; off >= 1; off >>= 1) ss += __shfl_xor(ss, off);
  const float r = rsqrtf(ss) * SCALE;
  v4h ha = { (_Float16)(a.x*r), (_Float16)(a.y*r), (_Float16)(a.z*r), (_Float16)(a.w*r) };
  v4h hb = { (_Float16)(b.x*r), (_Float16)(b.y*r), (_Float16)(b.z*r), (_Float16)(b.w*r) };
  v4h* o = (v4h*)(xn + (size_t)row * DIM);
  o[lane]      = ha;
  o[lane + 32] = hb;
}

// ---------------- epilogue helpers ------------------------------------------
__device__ __forceinline__ void epi_special(const v8f& d, v8f& rsum, v8f& psum,
                                            bool isdiag, bool ispos, int lrow, int s8) {
  #pragma unroll
  for (int v = 0; v < 8; ++v) {
    const bool hit = (lrow == v + s8);
    const float sim = d[v];               // = raw_sim / T * log2(e)
    const float ev  = fast_exp2(sim);
    rsum[v] += (isdiag && hit) ? 0.0f : ev;   // drop self-sim from denom
    psum[v] += (ispos  && hit) ? sim  : 0.0f; // capture positive-pair value
  }
}
__device__ __forceinline__ void epi_fast(const v8f& d, v8f& rsum) {
  #pragma unroll
  for (int v = 0; v < 8; ++v) rsum[v] += fast_exp2(d[v]);
}

// ---------------- Kernel 2: fused WMMA sims + exp2 + row-sum + pos ----------
// Grid: (TWO_N/512, SEGS). Block: 256 threads = 8 waves.
// Wave owns 64 rows as four 16-row strips; each LDS B-fragment feeds four
// independent WMMA accumulation chains (32 WMMAs per 8 KB of LDS reads).
__global__ __launch_bounds__(256) void _simclr_main(const _Float16* __restrict__ xn,
                                                    float* __restrict__ rs_part,
                                                    float* __restrict__ ps_part) {
  __shared__ __align__(16) _Float16 ldsB[2][16 * DIM];   // 2 x 8 KB double buffer

  const int tid  = threadIdx.x;
  const int lane = tid & 31;
  const int lrow = lane & 15;
  const int hi   = lane >> 4;      // 0 or 1 (lane half)
  const int s8   = hi << 3;        // 0 or 8

  const int R    = blockIdx.x * ROWS_PER_BLK;
  const int seg  = blockIdx.y;
  const int ct0  = seg * COL_TILES_PER_SEG;

  // Wave-uniform values as true scalars -> s_cmp/s_cbranch in the epilogue.
  const int rowbase = __builtin_amdgcn_readfirstlane(R + (tid >> 5) * ROWS_PER_WAVE);
  const int ctd = rowbase >> 4;                            // diag tiles: ctd..ctd+3
  const int ctp = ((rowbase + NHALF) & (TWO_N - 1)) >> 4;  // pos tiles:  ctp..ctp+3

  // Four resident A tile-sets (64 rows), full K=256.
  // ISA 16-bit A layout per 32-chunk: lanes 0-15 hold K {0..7,16..23},
  // lanes 16-31 hold K {8..15,24..31}.
  const v4u* arow = (const v4u*)xn + (size_t)(rowbase + lrow) * (DIM / 8);
  HFrag A[TILES_M][8];
  #pragma unroll
  for (int m = 0; m < TILES_M; ++m) {
    #pragma unroll
    for (int c = 0; c < 8; ++c) {
      A[m][c].u[0] = arow[m * 16 * (DIM / 8) + 4 * c + hi];
      A[m][c].u[1] = arow[m * 16 * (DIM / 8) + 4 * c + 2 + hi];
    }
  }

  const v8f vzero = {0.f,0.f,0.f,0.f,0.f,0.f,0.f,0.f};
  v8f rs[TILES_M], ps[TILES_M];
  #pragma unroll
  for (int m = 0; m < TILES_M; ++m) { rs[m] = vzero; ps[m] = vzero; }

  const unsigned lds0 = lds_low32(&ldsB[0][0]);
  const unsigned lds1 = lds_low32(&ldsB[1][0]);

  // Preload first B tile (16 rows of X^ = 8 KB contiguous) via async-to-LDS.
  {
    const v4u* g = (const v4u*)xn + (size_t)(ct0 * 16) * (DIM / 8);
    async_ld_b128(lds0 + tid * 16,        g + tid);
    async_ld_b128(lds0 + tid * 16 + 4096, g + tid + 256);
  }
  wait_async0();
  __syncthreads();

  for (int t = 0; t < COL_TILES_PER_SEG; ++t) {
    const int ct = ct0 + t;
    const v4u* cur = (const v4u*)(&ldsB[t & 1][0]);

    if (t + 1 < COL_TILES_PER_SEG) {     // async prefetch of next tile
      const unsigned nb = (t & 1) ? lds0 : lds1;
      const v4u* g = (const v4u*)xn + (size_t)((ct + 1) * 16) * (DIM / 8);
      async_ld_b128(nb + tid * 16,        g + tid);
      async_ld_b128(nb + tid * 16 + 4096, g + tid + 256);
    }

    // B fragments (dense 16-bit B 32x16: lanes 0-15 = K {0..7,16..23}, lanes 16-31 =
    // K {8..15,24..31} of each 32-chunk); each fragment feeds all four chains.
    const v4u* brow = cur + lrow * (DIM / 8);
    v8f d[TILES_M];
    #pragma unroll
    for (int m = 0; m < TILES_M; ++m) d[m] = vzero;
    #pragma unroll
    for (int c = 0; c < 8; ++c) {
      HFrag B;
      B.u[0] = brow[4 * c + hi];
      B.u[1] = brow[4 * c + 2 + hi];
      #pragma unroll
      for (int m = 0; m < TILES_M; ++m)
        d[m] = __builtin_amdgcn_wmma_f32_16x16x32_f16(false, A[m][c].h, false, B.h,
                                                      (short)0, d[m], false, false);
    }

    // Epilogue: diag/pos strips are scalar-uniform -> s_cbranch; fast path is
    // one v_exp_f32 + one v_add per element, nothing else.
    if ((unsigned)(ct - ctd) < TILES_M || (unsigned)(ct - ctp) < TILES_M) {
      #pragma unroll
      for (int m = 0; m < TILES_M; ++m)
        epi_special(d[m], rs[m], ps[m], ct == ctd + m, ct == ctp + m, lrow, s8);
    } else {
      #pragma unroll
      for (int m = 0; m < TILES_M; ++m) epi_fast(d[m], rs[m]);
    }

    wait_async0();     // prefetch landed in LDS
    __syncthreads();   // everyone done reading cur / writing nxt
  }

  // Reduce the 16 column-lanes per row, write per-(segment,row) partials.
  float* rsp = rs_part + (size_t)seg * TWO_N;
  float* psp = ps_part + (size_t)seg * TWO_N;
  #pragma unroll
  for (int m = 0; m < TILES_M; ++m) {
    #pragma unroll
    for (int v = 0; v < 8; ++v) {
      float a = rs[m][v], b = ps[m][v];
      #pragma unroll
      for (int off = 8; off >= 1; off >>= 1) {
        a += __shfl_xor(a, off);
        b += __shfl_xor(b, off);
      }
      if (lrow == 0) {
        rsp[rowbase + 16 * m + v + s8] = a;
        psp[rowbase + 16 * m + v + s8] = b;
      }
    }
  }
}

// ---------------- Kernel 3: deterministic combine + mean --------------------
__global__ __launch_bounds__(256) void _simclr_finalize(const float* __restrict__ rs_part,
                                                        const float* __restrict__ ps_part,
                                                        float* __restrict__ out) {
  __shared__ float sm[256];
  float acc = 0.f;
  for (int r = threadIdx.x; r < TWO_N; r += 256) {
    float rs = 0.f, ps = 0.f;
    #pragma unroll
    for (int sg = 0; sg < SEGS; ++sg) {
      rs += rs_part[sg * TWO_N + r];
      ps += ps_part[sg * TWO_N + r];
    }
    acc += __logf(rs) - ps * LN2;     // log(denom) - sim_pos / T
  }
  sm[threadIdx.x] = acc;
  __syncthreads();
  for (int st = 128; st > 0; st >>= 1) {
    if (threadIdx.x < st) sm[threadIdx.x] += sm[threadIdx.x + st];
    __syncthreads();
  }
  if (threadIdx.x == 0) out[0] = sm[0] * (1.0f / TWO_N);
}

extern "C" void kernel_launch(void* const* d_in, const int* in_sizes, int n_in,
                              void* d_out, int out_size, void* d_ws, size_t ws_size,
                              hipStream_t stream) {
  (void)in_sizes; (void)n_in; (void)out_size; (void)ws_size;
  const float* x = (const float*)d_in[0];
  float* out = (float*)d_out;

  // Workspace: [X^ fp16: 4 MB][rowsum partials: 512 KB][pos partials: 512 KB]
  _Float16* xn = (_Float16*)d_ws;
  float* rs_part = (float*)((char*)d_ws + (size_t)TWO_N * DIM * sizeof(_Float16));
  float* ps_part = rs_part + (size_t)SEGS * TWO_N;

  _simclr_norm<<<dim3(TWO_N / 8), dim3(256), 0, stream>>>(x, xn);
  _simclr_main<<<dim3(TWO_N / ROWS_PER_BLK, SEGS), dim3(256), 0, stream>>>(xn, rs_part, ps_part);
  _simclr_finalize<<<dim3(1), dim3(256), 0, stream>>>(rs_part, ps_part, out);
}
